// MoeLayer_28286654611480
// MI455X (gfx1250) — matrix operand order repaired and checked by reading, not compiled, not run on hardware
//
#include <hip/hip_runtime.h>
#include <hip/hip_bf16.h>

// ---------------------------------------------------------------------------
// MoE layer for MI455X (gfx1250, wave32, WMMA).
// 275 GFLOP vs ~320MB unique bytes -> compute bound once tokens are grouped
// per expert (avg 2048 tokens/expert => ~128x weight-tile reuse). Strategy:
//   * one-time per-call swizzle: w1/w2 fp32 -> bf16 in B-fragment register
//     order (bf16 weights = 128MB, fits the 192MB L2)
//   * x tile staged into LDS in A-fragment order via GLOBAL_LOAD_ASYNC_TO_LDS
//     (ASYNCcnt path) when available; h kept in LDS (160KB dynamic LDS)
//   * hot loop = 2x global_load_b128 + 2x ds_load_b128 + v_wmma_f32_16x16x32_bf16
//   * branchless tanh-gelu: v*sigmoid(2u) via v_exp_f32 + v_rcp_f32
// ---------------------------------------------------------------------------

typedef __attribute__((ext_vector_type(16))) __bf16    v16bf;
typedef __attribute__((ext_vector_type(8)))  float     v8f;
typedef __attribute__((ext_vector_type(4)))  unsigned  v4u;
typedef __attribute__((ext_vector_type(4)))  int       v4i;

#define Bc  4
#define Sc  2048
#define Dc  1024
#define Ec  8
#define Hc  4096
#define BSc (Bc * Sc)

#define KS1 (Dc / 32)   // 32  k-steps, phase 1
#define NT1 (Hc / 16)   // 256 n-tiles, phase 1
#define KS2 (Hc / 32)   // 128 k-steps, phase 2
#define NT2 (Dc / 16)   // 64  n-tiles, phase 2

#define HAS_ASYNC_LDS (__has_builtin(__builtin_amdgcn_global_load_async_to_lds_b128) && \
                       __has_builtin(__builtin_amdgcn_s_wait_asynccnt))

__device__ __forceinline__ unsigned short f2bf(float f) {
  return __builtin_bit_cast(unsigned short, (__bf16)f);   // hw cvt, RNE
}
__device__ __forceinline__ unsigned pack_bf16(float a, float b) {
  return (unsigned)f2bf(a) | ((unsigned)f2bf(b) << 16);
}

// branchless sigmoid: 1/(1+exp(-t)); exact at +-inf, no EXEC-mask branches
__device__ __forceinline__ float fast_sigmoid(float t) {
#if __has_builtin(__builtin_amdgcn_exp2f) && __has_builtin(__builtin_amdgcn_rcpf)
  return __builtin_amdgcn_rcpf(1.0f + __builtin_amdgcn_exp2f(-1.4426950408889634f * t));
#else
  return 1.0f / (1.0f + __expf(-t));
#endif
}
// tanh-gelu (jax default): 0.5*v*(1+tanh(u)) == v*sigmoid(2u)
__device__ __forceinline__ float gelu_tanh(float v) {
  float u = 0.7978845608028654f * (v + 0.044715f * v * v * v);
  return v * fast_sigmoid(2.0f * u);
}

__global__ void zero_out_kernel(float* __restrict__ out, int n, int* __restrict__ counts) {
  int i = blockIdx.x * blockDim.x + threadIdx.x;
  if (i < n) out[i] = 0.0f;
  if (i < Ec) counts[i] = 0;
}

__global__ void convert_x_bf16(const float* __restrict__ x,
                               unsigned short* __restrict__ xb, int n) {
  int i = blockIdx.x * blockDim.x + threadIdx.x;
  if (i < n) xb[i] = f2bf(x[i]);
}

// fp32 [E][K][N] row-major -> bf16 tiles in B-fragment order.
// Tile (e,nt,ks) = 256 dwords; dword (l,j) packs w[k][n], w[k+1][n] with
// n = nt*16 + (l&15), k = ks*32 + (l>=16 ? 8 : 0) + (j>=4 ? 16 : 0) + 2(j&3).   (ISA 7.12.2)
__global__ void swizzle_w_bf16(const float* __restrict__ w, unsigned* __restrict__ o,
                               int KS, int NT, int NTOT, int ndwords) {
  int idx = blockIdx.x * blockDim.x + threadIdx.x;
  if (idx >= ndwords) return;
  int j  = idx & 7;
  int l  = (idx >> 3) & 31;
  int t  = idx >> 8;
  int ks = t % KS;
  int nt = (t / KS) % NT;
  int e  = t / (KS * NT);
  int n  = nt * 16 + (l & 15);
  int k  = ks * 32 + ((l >= 16) ? 8 : 0) + ((j >= 4) ? 16 : 0) + 2 * (j & 3);
  const float* we = w + (size_t)e * (size_t)(KS * 32) * NTOT;
  o[idx] = pack_bf16(we[(size_t)k * NTOT + n], we[(size_t)(k + 1) * NTOT + n]);
}

// Gate: logits = x @ gate_w, top-2, softmax over the 2, compact expert lists.
__global__ void gate_route(const float* __restrict__ x, const float* __restrict__ gw,
                           int* __restrict__ counts, int* __restrict__ tokIdx,
                           float* __restrict__ tokCoef) {
  int t = blockIdx.x * blockDim.x + threadIdx.x;
  if (t >= BSc) return;
  float lg[Ec];
#pragma unroll
  for (int e = 0; e < Ec; ++e) lg[e] = 0.0f;
  const float* xr = x + (size_t)t * Dc;
  for (int d = 0; d < Dc; ++d) {
    float xv = xr[d];
#pragma unroll
    for (int e = 0; e < Ec; ++e) lg[e] += xv * gw[d * Ec + e];
  }
  int e0 = 0; float l0 = lg[0];
#pragma unroll
  for (int e = 1; e < Ec; ++e) { if (lg[e] > l0) { l0 = lg[e]; e0 = e; } }
  int e1 = (e0 == 0) ? 1 : 0; float l1 = lg[e1];
#pragma unroll
  for (int e = 0; e < Ec; ++e) { if (e != e0 && lg[e] > l1) { l1 = lg[e]; e1 = e; } }
  float z  = __expf(l1 - l0);
  float c0 = 1.0f / (1.0f + z);
  float c1 = z * c0;
  int p0 = atomicAdd(&counts[e0], 1);
  tokIdx[e0 * BSc + p0] = t;  tokCoef[e0 * BSc + p0] = c0;
  int p1 = atomicAdd(&counts[e1], 1);
  tokIdx[e1 * BSc + p1] = t;  tokCoef[e1 * BSc + p1] = c1;
}

// A-fragment halfword address for element (row m, col k) in an LDS fragment
// buffer laid out [kstep][lane][8 dwords]  (ISA 16-bit A 16x32 layout).
__device__ __forceinline__ void afrag_pos(int m, int k, int& dword, int& pos) {
  int ks  = k >> 5, kk = k & 31;
  int lhk = ((kk & 15) >= 8) ? 1 : 0;
  int j   = ((kk >= 16) ? 4 : 0) + ((kk & 7) >> 1);
  dword   = (ks * 32 + (m + 16 * lhk)) * 8 + j;
  pos     = kk & 1;
}

__global__ __launch_bounds__(256, 1)
void moe_expert_gemm(const unsigned short* __restrict__ xb,
                     const v4u* __restrict__ w1s, const float* __restrict__ b1,
                     const v4u* __restrict__ w2s, const float* __restrict__ b2,
                     const int* __restrict__ counts, const int* __restrict__ tokIdx,
                     const float* __restrict__ tokCoef, float* __restrict__ out) {
  const int e    = blockIdx.y;
  const int mblk = blockIdx.x;
  const int cnt  = counts[e];
  if (mblk * 16 >= cnt) return;

  extern __shared__ char smem[];
  unsigned* xsf = (unsigned*)smem;        // [KS1][32][8] dwords = 32KB (x A-frags)
  unsigned* hsf = xsf + KS1 * 32 * 8;     // [KS2][32][8] dwords = 128KB (h A-frags)
  __shared__ int   s_tok[16];
  __shared__ float s_coef[16];

  const int tid = threadIdx.x;
  if (tid < 16) {
    int r = mblk * 16 + tid;
    if (r < cnt) { s_tok[tid] = tokIdx[e * BSc + r]; s_coef[tid] = tokCoef[e * BSc + r]; }
    else         { s_tok[tid] = 0;                   s_coef[tid] = 0.0f; }
  }
  __syncthreads();

  // Stage x tile into A-fragment order. An 8-aligned group of 8 bf16 (16B in
  // global) lands in 4 *contiguous* fragment dwords (16B in LDS), so each
  // chunk is one 128-bit transfer -> async-to-LDS (ASYNCcnt) when available.
  {
    // chunk c (of 2048): row m = c>>7, k = (c&127)*8
    for (int c = tid; c < 16 * (Dc / 8); c += 256) {
      int m  = c >> 7;
      int k  = (c & 127) * 8;
      int ks = k >> 5, kk = k & 31;
      int lhk   = ((kk & 15) >= 8) ? 1 : 0;
      int jbase = (kk >= 16) ? 4 : 0;
      unsigned* ldst = &xsf[(ks * 32 + (m + 16 * lhk)) * 8 + jbase];
      const unsigned short* gsrc = xb + (size_t)s_tok[m] * Dc + k;
#if HAS_ASYNC_LDS
      __builtin_amdgcn_global_load_async_to_lds_b128(
          (__attribute__((address_space(1))) v4i*)gsrc,
          (__attribute__((address_space(3))) v4i*)ldst, 0, 0);
#else
      *(v4u*)ldst = *(const v4u*)gsrc;
#endif
    }
#if HAS_ASYNC_LDS
    __builtin_amdgcn_s_wait_asynccnt(0);
#endif
  }
  __syncthreads();

  const int wave = tid >> 5;
  const int lane = tid & 31;
  const int lm   = lane & 15;
  const int lhi  = lane >> 4;

  // ---------------- Phase 1: h = gelu(x @ w1[e] + b1[e]) --------------------
  const float* b1e = b1 + (size_t)e * Hc;
  for (int nt0 = 0; nt0 < NT1 / 8; ++nt0) {          // 32 n-tiles per wave
    const int nt   = wave * (NT1 / 8) + nt0;
    const int ncol = nt * 16 + lm;
    const v4u*      bt = w1s + ((size_t)(e * NT1 + nt) * KS1) * 64 + lane * 2;
    const unsigned* at = xsf + lane * 8;
    v8f acc = {0.f, 0.f, 0.f, 0.f, 0.f, 0.f, 0.f, 0.f};
#pragma unroll 4
    for (int ks = 0; ks < KS1; ++ks) {
      __builtin_prefetch((const void*)(bt + 64), 0, 1);  // next B tile (global_prefetch_b8)
      union { v4u q[2]; v16bf v; } A, B;
      A.q[0] = ((const v4u*)at)[0];  A.q[1] = ((const v4u*)at)[1];   // 2x ds_load_b128
      B.q[0] = bt[0];                B.q[1] = bt[1];                 // 2x global_load_b128
      acc = __builtin_amdgcn_wmma_f32_16x16x32_bf16(false, A.v, false, B.v,
                                                    (short)0, acc, false, false);
      at += 32 * 8;
      bt += 64;
    }
    // bias + gelu, park h in phase-2 A-fragment order
    const float bias = b1e[ncol];
    int dwb, pos;
    afrag_pos(0, ncol, dwb, pos);                      // row offset added below
#pragma unroll
    for (int r = 0; r < 8; ++r) {                      // C/D: M = r + 8*lhi
      int   m = r + lhi * 8;
      float g = gelu_tanh(acc[r] + bias);
      ((unsigned short*)&hsf[dwb + m * 8])[pos] = f2bf(g);
    }
  }
  __syncthreads();

  // ---------------- Phase 2: y = h @ w2[e] + b2[e]; out += coef*y -----------
  const float* b2e = b2 + (size_t)e * Dc;
  for (int nt0 = 0; nt0 < NT2 / 8; ++nt0) {           // 8 n-tiles per wave
    const int nt   = wave * (NT2 / 8) + nt0;
    const int ncol = nt * 16 + lm;
    const v4u*      bt = w2s + ((size_t)(e * NT2 + nt) * KS2) * 64 + lane * 2;
    const unsigned* at = hsf + lane * 8;
    v8f acc = {0.f, 0.f, 0.f, 0.f, 0.f, 0.f, 0.f, 0.f};
#pragma unroll 4
    for (int ks = 0; ks < KS2; ++ks) {
      __builtin_prefetch((const void*)(bt + 64), 0, 1);
      union { v4u q[2]; v16bf v; } A, B;
      A.q[0] = ((const v4u*)at)[0];  A.q[1] = ((const v4u*)at)[1];
      B.q[0] = bt[0];                B.q[1] = bt[1];
      acc = __builtin_amdgcn_wmma_f32_16x16x32_bf16(false, A.v, false, B.v,
                                                    (short)0, acc, false, false);
      at += 32 * 8;
      bt += 64;
    }
    const float bias = b2e[ncol];
#pragma unroll
    for (int r = 0; r < 8; ++r) {
      int   m = r + lhi * 8;
      float y = acc[r] + bias;
      atomicAdd(out + (size_t)s_tok[m] * Dc + ncol, s_coef[m] * y);
    }
  }
}

extern "C" void kernel_launch(void* const* d_in, const int* in_sizes, int n_in,
                              void* d_out, int out_size, void* d_ws, size_t ws_size,
                              hipStream_t stream) {
  (void)in_sizes; (void)n_in; (void)out_size; (void)ws_size;
  const float* x      = (const float*)d_in[0];
  const float* gate_w = (const float*)d_in[1];
  const float* w1     = (const float*)d_in[2];
  const float* b1     = (const float*)d_in[3];
  const float* w2     = (const float*)d_in[4];
  const float* b2     = (const float*)d_in[5];
  float* out = (float*)d_out;

  // ws: xb (16MB) | w1s (64MB) | w2s (64MB) | counts | tokIdx | tokCoef
  unsigned short* xb  = (unsigned short*)d_ws;
  unsigned* w1s = (unsigned*)(xb + (size_t)BSc * Dc);
  unsigned* w2s = w1s + (size_t)Ec * Dc * Hc / 2;
  int*   counts  = (int*)(w2s + (size_t)Ec * Hc * Dc / 2);
  int*   tokIdx  = counts + 32;
  float* tokCoef = (float*)(tokIdx + Ec * BSc);

  const int ntot = BSc * Dc;
  zero_out_kernel<<<(ntot + 255) / 256, 256, 0, stream>>>(out, ntot, counts);
  convert_x_bf16 <<<(ntot + 255) / 256, 256, 0, stream>>>(x, xb, ntot);

  const int nw = Ec * Dc * Hc / 2;   // dwords per weight tensor
  swizzle_w_bf16<<<(nw + 255) / 256, 256, 0, stream>>>(w1, w1s, KS1, NT1, Hc, nw);
  swizzle_w_bf16<<<(nw + 255) / 256, 256, 0, stream>>>(w2, w2s, KS2, NT2, Dc, nw);

  gate_route<<<(BSc + 255) / 256, 256, 0, stream>>>(x, gate_w, counts, tokIdx, tokCoef);

  dim3 grid(BSc / 16, Ec);                              // 512 m-blocks x 8 experts
  const size_t lds = (size_t)(KS1 + KS2) * 32 * 8 * 4;  // 160 KB dynamic LDS
  moe_expert_gemm<<<grid, 256, lds, stream>>>(xb, (const v4u*)w1s, b1,
                                              (const v4u*)w2s, b2,
                                              counts, tokIdx, tokCoef, out);
}
